// GraphTransformerLayer_80633716015119
// MI455X (gfx1250) — matrix operand order repaired and checked by reading, not compiled, not run on hardware
//
#include <hip/hip_runtime.h>
#include <hip/hip_bf16.h>
#include <math.h>

typedef __attribute__((ext_vector_type(16))) __bf16 v16bf;
typedef __attribute__((ext_vector_type(8)))  __bf16 v8bf;
typedef __attribute__((ext_vector_type(8)))  float  v8f;
typedef __attribute__((ext_vector_type(4)))  float  v4f;

// ---------------------------------------------------------------------------
// Tensor Data Mover support (gfx1250).  5-arg builtin on ROCm/clang-22,
// 6-arg on amdgpu-toolchain (which ships the TDM header) — hedge on both.
// ---------------------------------------------------------------------------
#if __has_builtin(__builtin_amdgcn_tensor_load_to_lds) && \
    __has_builtin(__builtin_amdgcn_s_wait_tensorcnt)
#define USE_TDM 1
#else
#define USE_TDM 0
#endif

#if USE_TDM
typedef unsigned int v4u  __attribute__((ext_vector_type(4)));
typedef int          v4i_t __attribute__((ext_vector_type(4)));
typedef int          v8i_t __attribute__((ext_vector_type(8)));

#if __has_include(<hip/amd_detail/amd_gfx1250_TDM.h>)
#define TDM_CALL(g0, g1, g2, g3) \
    __builtin_amdgcn_tensor_load_to_lds(g0, g1, g2, g3, (v8i_t){0,0,0,0,0,0,0,0}, 0)
#else
#define TDM_CALL(g0, g1, g2, g3) \
    __builtin_amdgcn_tensor_load_to_lds(g0, g1, g2, g3, 0)
#endif

// 2D bf16 tile load: tile 32(K) x 64(M rows), row stride = K elements,
// pad_enable: after every 64 bytes (16 dwords, code 3) insert 16 bytes
// (4 dwords, code 3) -> LDS rows of 40 bf16 (bank-conflict padding in HW).
// OOB rows/cols (beyond avail_*) read as zero -> handles the partial M tile.
__device__ inline void tdm_load_tile_a(unsigned lds_off, const void* gp,
                                       int avail_k, int avail_m,
                                       int row_stride_elems) {
    unsigned long long ga = (unsigned long long)gp;
    v4u g0;
    g0.x = 1u;                                   // count = 1 valid descriptor
    g0.y = lds_off;                              // lds_addr
    g0.z = (unsigned)ga;                         // global_addr[31:0]
    g0.w = (unsigned)((ga >> 32) & 0x01ffffffu) | (2u << 30);  // addr hi | type=2
    v8i_t g1;
    g1[0] = (int)((1u << 16) |                   // data_size = 2 bytes
                  (1u << 20) |                   // pad_enable
                  (3u << 22) |                   // pad_interval: 16 dwords
                  (3u << 25));                   // pad_amount:   4 dwords
    g1[1] = (int)(((unsigned)avail_k & 0xffffu) << 16);                 // td0 lo
    g1[2] = (int)((((unsigned)avail_k >> 16) & 0xffffu) |
                  (((unsigned)avail_m & 0xffffu) << 16));               // td0 hi|td1 lo
    g1[3] = (int)((((unsigned)avail_m >> 16) & 0xffffu) | (32u << 16)); // td1 hi|tile0=32
    g1[4] = 64;                                  // tile_dim1 = 64, tile_dim2 = 0
    g1[5] = row_stride_elems;                    // tensor_dim0_stride lo
    g1[6] = 0;
    g1[7] = 0;
    v4i_t z = {0, 0, 0, 0};
    TDM_CALL(g0, g1, z, z);
}

// Transposing B-tile load via 3D descriptor:
//   tile (x=1, y=32 K-rows stride=Nc, z=128 cols stride=1)
// LDS receives elements column-major: sBt[col][k], i.e. a 128x32 transposed
// image of the 32x128 row-major global tile. No padding (tile_dim0 = 2B).
__device__ inline void tdm_load_tile_b_t(unsigned lds_off, const void* gp,
                                         int avail_k, int avail_n,
                                         int row_stride_elems) {
    unsigned long long ga = (unsigned long long)gp;
    v4u g0;
    g0.x = 1u;
    g0.y = lds_off;
    g0.z = (unsigned)ga;
    g0.w = (unsigned)((ga >> 32) & 0x01ffffffu) | (2u << 30);
    v8i_t g1;
    g1[0] = (int)(1u << 16);                     // data_size = 2 bytes
    g1[1] = (int)(1u << 16);                     // tensor_dim0 = 1
    g1[2] = (int)(((unsigned)avail_k & 0xffffu) << 16);                 // tensor_dim1 lo
    g1[3] = (int)((((unsigned)avail_k >> 16) & 0xffffu) | (1u << 16));  // td1 hi|tile0=1
    g1[4] = (int)(32u | (128u << 16));           // tile_dim1=32, tile_dim2=128
    g1[5] = row_stride_elems;                    // dim0_stride = Nc (walk K)
    g1[6] = (int)(1u << 16);                     // dim1_stride = 1  (walk cols)
    g1[7] = 0;
    v4i_t g2 = {0, 0, 0, 0};
    g2[0] = avail_n;                             // tensor_dim2 (OOB cols -> 0)
    v4i_t z = {0, 0, 0, 0};
    TDM_CALL(g0, g1, g2, z);
}
#endif  // USE_TDM

// ---------------------------------------------------------------------------
// float -> bf16 conversion (weights)
// ---------------------------------------------------------------------------
__global__ void f32_to_bf16_kernel(const float* __restrict__ in,
                                   __bf16* __restrict__ out, int n) {
    int i = (blockIdx.x * 256 + threadIdx.x) * 4;
    if (i + 3 < n) {
        v4f v = *(const v4f*)(in + i);
        out[i + 0] = (__bf16)v[0];
        out[i + 1] = (__bf16)v[1];
        out[i + 2] = (__bf16)v[2];
        out[i + 3] = (__bf16)v[3];
    } else {
        for (int j = i; j < n; ++j) out[j] = (__bf16)in[j];
    }
}

// ---------------------------------------------------------------------------
// LayerNorm over D=512, one wave32 per row, bf16 output
// ---------------------------------------------------------------------------
__global__ __launch_bounds__(256) void ln_kernel(const float* __restrict__ x,
                                                 const float* __restrict__ g,
                                                 const float* __restrict__ b,
                                                 __bf16* __restrict__ out,
                                                 int nrows) {
    int wid  = threadIdx.x >> 5;
    int lane = threadIdx.x & 31;
    int row  = blockIdx.x * 8 + wid;
    if (row >= nrows) return;
    const float* xr = x + (size_t)row * 512;
    int base = lane * 16;
    float vals[16];
    float sum = 0.f, sq = 0.f;
#pragma unroll
    for (int j = 0; j < 16; j += 4) {
        v4f v = *(const v4f*)(xr + base + j);
#pragma unroll
        for (int t = 0; t < 4; ++t) {
            vals[j + t] = v[t];
            sum += v[t];
            sq  += v[t] * v[t];
        }
    }
#pragma unroll
    for (int m = 16; m >= 1; m >>= 1) {
        sum += __shfl_xor(sum, m, 32);
        sq  += __shfl_xor(sq,  m, 32);
    }
    float mu  = sum * (1.f / 512.f);
    float var = sq * (1.f / 512.f) - mu * mu;
    float inv = rsqrtf(var + 1e-5f);
    __bf16* outr = out + (size_t)row * 512;
#pragma unroll
    for (int j = 0; j < 16; ++j) {
        float o = (vals[j] - mu) * inv * g[base + j] + b[base + j];
        outr[base + j] = (__bf16)o;
    }
}

// ---------------------------------------------------------------------------
// WMMA bf16 GEMM:  C[M,Nc] = A[M,K] * B[K,Nc] + bias (+resid) (gelu) (cast)
// Block tile 64x128, 8 waves, each wave = 16x64 (4x v_wmma_f32_16x16x32_bf16)
// Tiles staged to LDS by the Tensor Data Mover (wave 0 issues descriptors,
// B is transposed by the DMA); s_wait_tensorcnt + workgroup barrier sync.
// ---------------------------------------------------------------------------
template <bool AF32, bool GELU_, bool OUTBF16>
__global__ __launch_bounds__(256) void gemm_wmma_kernel(
    const void* __restrict__ Aq, const __bf16* __restrict__ B,
    const float* __restrict__ bias, const float* __restrict__ resid,
    void* __restrict__ Cq, int M, int K, int Nc) {
    __shared__ __bf16 sA[64][40];    // 64 rows x 32 K (+8 pad, TDM pad feature)
    __shared__ __bf16 sBt[128][32];  // transposed: 128 cols x 32 K

    const int tid   = threadIdx.x;
    const int lane  = tid & 31;
    const int wid   = tid >> 5;
    const int waveM = wid & 3;   // 4 row-groups of 16
    const int waveN = wid >> 2;  // 2 col-groups of 64
    const int m0 = blockIdx.y * 64;
    const int n0 = blockIdx.x * 128;

    v8f acc[4];
#pragma unroll
    for (int i = 0; i < 4; ++i)
#pragma unroll
        for (int j = 0; j < 8; ++j) acc[i][j] = 0.f;

    const int arow = tid >> 2;          // 0..63
    const int acol = (tid & 3) * 8;     // 0,8,16,24
#if !USE_TDM
    const int brow = tid >> 3;          // 0..31
    const int bcol = (tid & 7) * 16;    // 0..112
#endif

    for (int k0 = 0; k0 < K; k0 += 32) {
#if USE_TDM
        if (tid < 32) {  // wave 0 drives the DMA
            tdm_load_tile_b_t((unsigned)(size_t)&sBt[0][0],
                              B + (size_t)k0 * Nc + n0, K - k0, Nc - n0, Nc);
            if (!AF32) {
                tdm_load_tile_a((unsigned)(size_t)&sA[0][0],
                                (const __bf16*)Aq + (size_t)m0 * K + k0,
                                K - k0, M - m0, K);
            }
        }
        if (AF32) {  // fp32 A (post-atomics): stage + convert manually
            int gr = m0 + arow;
            v8bf tmp;
            if (gr < M) {
                const float* Af = (const float*)Aq;
                v4f a0 = *(const v4f*)(Af + (size_t)gr * K + k0 + acol);
                v4f a1 = *(const v4f*)(Af + (size_t)gr * K + k0 + acol + 4);
#pragma unroll
                for (int j = 0; j < 4; ++j) {
                    tmp[j]     = (__bf16)a0[j];
                    tmp[4 + j] = (__bf16)a1[j];
                }
            } else {
#pragma unroll
                for (int j = 0; j < 8; ++j) tmp[j] = (__bf16)0.f;
            }
            *(v8bf*)&sA[arow][acol] = tmp;
        }
        if (tid < 32) __builtin_amdgcn_s_wait_tensorcnt((short)0);
#else
        // -------- fallback: manual staging --------
        {
            int gr = m0 + arow;
            v8bf tmp;
            if (gr < M) {
                if (AF32) {
                    const float* Af = (const float*)Aq;
                    v4f a0 = *(const v4f*)(Af + (size_t)gr * K + k0 + acol);
                    v4f a1 = *(const v4f*)(Af + (size_t)gr * K + k0 + acol + 4);
#pragma unroll
                    for (int j = 0; j < 4; ++j) {
                        tmp[j]     = (__bf16)a0[j];
                        tmp[4 + j] = (__bf16)a1[j];
                    }
                } else {
                    const __bf16* Ab = (const __bf16*)Aq;
                    tmp = *(const v8bf*)(Ab + (size_t)gr * K + k0 + acol);
                }
            } else {
#pragma unroll
                for (int j = 0; j < 8; ++j) tmp[j] = (__bf16)0.f;
            }
            *(v8bf*)&sA[arow][acol] = tmp;
        }
        {
            const __bf16* Bp = B + (size_t)(k0 + brow) * Nc + n0 + bcol;
            v8bf b0 = *(const v8bf*)Bp;
            v8bf b1 = *(const v8bf*)(Bp + 8);
#pragma unroll
            for (int j = 0; j < 8; ++j) {
                sBt[bcol + j][brow]     = b0[j];
                sBt[bcol + 8 + j][brow] = b1[j];
            }
        }
#endif
        __syncthreads();

        // ---- WMMA compute: fragments per ISA 16-bit layouts ----
        const int r   = lane & 15;
        const int kb  = (lane < 16) ? 0 : 8;   // A half-lane K base
        const int kbb = (lane < 16) ? 0 : 8;   // B half-lane K base
        v8bf alo = *(const v8bf*)&sA[waveM * 16 + r][kb];
        v8bf ahi = *(const v8bf*)&sA[waveM * 16 + r][kb + 16];
        v16bf afrag = __builtin_shufflevector(alo, ahi, 0, 1, 2, 3, 4, 5, 6, 7,
                                              8, 9, 10, 11, 12, 13, 14, 15);
#pragma unroll
        for (int nt = 0; nt < 4; ++nt) {
            const __bf16* bp = &sBt[waveN * 64 + nt * 16 + r][kbb];
            v8bf blo = *(const v8bf*)bp;
            v8bf bhi = *(const v8bf*)(bp + 16);
            v16bf bfrag = __builtin_shufflevector(blo, bhi, 0, 1, 2, 3, 4, 5, 6,
                                                  7, 8, 9, 10, 11, 12, 13, 14, 15);
            acc[nt] = __builtin_amdgcn_wmma_f32_16x16x32_bf16(
                false, afrag, false, bfrag, (short)0, acc[nt], false, false);
        }
        __syncthreads();
    }

    // ---- epilogue: bias (+gelu) (+resid), cast, store ----
#pragma unroll
    for (int nt = 0; nt < 4; ++nt) {
        int gcol = n0 + waveN * 64 + nt * 16 + (lane & 15);
        float bv = bias[gcol];
#pragma unroll
        for (int i = 0; i < 8; ++i) {
            int grow = m0 + waveM * 16 + ((lane < 16) ? i : i + 8);
            if (grow < M) {
                float v = acc[nt][i] + bv;
                if (GELU_) v = 0.5f * v * (1.f + erff(v * 0.70710678118f));
                if (resid) v += resid[(size_t)grow * Nc + gcol];
                if (OUTBF16)
                    ((__bf16*)Cq)[(size_t)grow * Nc + gcol] = (__bf16)v;
                else
                    ((float*)Cq)[(size_t)grow * Nc + gcol] = v;
            }
        }
    }
}

// ---------------------------------------------------------------------------
// Edge phase
// ---------------------------------------------------------------------------
__device__ inline void atomicMaxF(float* addr, float val) {
    if (val >= 0.f)
        atomicMax((int*)addr, __float_as_int(val));
    else
        atomicMin((unsigned int*)addr, (unsigned int)__float_as_int(val));
}

__global__ void init_kernel(float* __restrict__ mx, float* __restrict__ s,
                            float* __restrict__ agg, int nh, int nd) {
    int i = blockIdx.x * 256 + threadIdx.x;
    if (i < nh) {
        mx[i] = -__builtin_inff();
        s[i]  = 0.f;
    }
    if (i < nd) agg[i] = 0.f;
}

// one wave32 per edge: 4 lanes per head dot q[src]·k[dst] over DH=64 (bf16)
__global__ __launch_bounds__(256) void edge_logits_kernel(
    const __bf16* __restrict__ qkv, const int* __restrict__ src,
    const int* __restrict__ dst, const float* __restrict__ dattn,
    const float* __restrict__ pattn, float* __restrict__ a,
    float* __restrict__ mx, int E) {
    int wid  = threadIdx.x >> 5;
    int lane = threadIdx.x & 31;
    int e = blockIdx.x * 8 + wid;
    if (e >= E) return;
    int sn = src[e], dn = dst[e];
    int head = lane >> 2;
    int off  = head * 64 + (lane & 3) * 16;
    const __bf16* qp = qkv + (size_t)sn * 1536 + off;          // q block
    const __bf16* kp = qkv + (size_t)dn * 1536 + 512 + off;    // k block
    v8bf q0 = *(const v8bf*)qp, q1 = *(const v8bf*)(qp + 8);
    v8bf k0 = *(const v8bf*)kp, k1 = *(const v8bf*)(kp + 8);
    float acc = 0.f;
#pragma unroll
    for (int j = 0; j < 8; ++j)
        acc += (float)q0[j] * (float)k0[j] + (float)q1[j] * (float)k1[j];
    acc += __shfl_xor(acc, 1, 32);
    acc += __shfl_xor(acc, 2, 32);
    if ((lane & 3) == 0) {
        // q scale D^-0.5 folded into the logit
        float aval = acc * 0.04419417382415922f + dattn[(size_t)e * 8 + head] +
                     pattn[(size_t)e * 8 + head];
        a[(size_t)e * 8 + head] = aval;
        atomicMaxF(&mx[(size_t)dn * 8 + head], aval);
    }
}

__global__ void edge_exp_kernel(float* __restrict__ a,
                                const int* __restrict__ dst,
                                const float* __restrict__ mx,
                                float* __restrict__ s, int EH) {
    int i = blockIdx.x * 256 + threadIdx.x;
    if (i >= EH) return;
    int e = i >> 3, h = i & 7;
    int d = dst[e];
    float ex = expf(a[i] - mx[d * 8 + h]);
    a[i] = ex;
    atomicAdd(&s[d * 8 + h], ex);
}

// one wave32 per edge: scatter v[src]*softmax into agg[dst] (f32 atomics, L2)
__global__ __launch_bounds__(256) void edge_agg_kernel(
    const __bf16* __restrict__ qkv, const int* __restrict__ src,
    const int* __restrict__ dst, const float* __restrict__ a,
    const float* __restrict__ s, float* __restrict__ agg, int E) {
    int wid  = threadIdx.x >> 5;
    int lane = threadIdx.x & 31;
    int e = blockIdx.x * 8 + wid;
    if (e >= E) return;
    int sn = src[e], dn = dst[e];
    int head = lane >> 2;  // 16 contiguous elems per lane -> head = lane/4
    float sa = a[(size_t)e * 8 + head] / s[(size_t)dn * 8 + head];
    const __bf16* vp = qkv + (size_t)sn * 1536 + 1024 + lane * 16;
    float* ap = agg + (size_t)dn * 512 + lane * 16;
    v8bf v0 = *(const v8bf*)vp, v1 = *(const v8bf*)(vp + 8);
#pragma unroll
    for (int j = 0; j < 8; ++j) {
        atomicAdd(ap + j,     (float)v0[j] * sa);
        atomicAdd(ap + 8 + j, (float)v1[j] * sa);
    }
}

// ---------------------------------------------------------------------------
// Launcher
// ---------------------------------------------------------------------------
extern "C" void kernel_launch(void* const* d_in, const int* in_sizes, int n_in,
                              void* d_out, int out_size, void* d_ws,
                              size_t ws_size, hipStream_t stream) {
    const float* node  = (const float*)d_in[0];
    const int*   src   = (const int*)d_in[1];
    const int*   dst   = (const int*)d_in[2];
    const float* dattn = (const float*)d_in[3];
    const float* pattn = (const float*)d_in[4];
    const float* Wqkv  = (const float*)d_in[5];
    const float* bqkv  = (const float*)d_in[6];
    const float* Wo    = (const float*)d_in[7];
    const float* bo    = (const float*)d_in[8];
    const float* ln1g  = (const float*)d_in[9];
    const float* ln1b  = (const float*)d_in[10];
    const float* W1    = (const float*)d_in[11];
    const float* b1    = (const float*)d_in[12];
    const float* W2    = (const float*)d_in[13];
    const float* b2    = (const float*)d_in[14];
    const float* ln2g  = (const float*)d_in[15];
    const float* ln2b  = (const float*)d_in[16];

    const int N = in_sizes[0] / 512;
    const int E = in_sizes[1];

    char* ws = (char*)d_ws;
    size_t off = 0;
    auto alloc = [&](size_t bytes) -> char* {
        char* p = ws + off;
        off += (bytes + 255) & ~(size_t)255;
        return p;
    };
    __bf16* qkv_b  = (__bf16*)alloc((size_t)N * 1536 * 2);
    __bf16* h_b    = (__bf16*)alloc((size_t)N * 512 * 2);   // LN1 out, reused for LN2 out
    __bf16* t_b    = (__bf16*)alloc((size_t)N * 2048 * 2);  // FFN hidden
    float*  h1     = (float*)alloc((size_t)N * 512 * 4);
    float*  agg    = (float*)alloc((size_t)N * 512 * 4);
    float*  abuf   = (float*)alloc((size_t)E * 8 * 4);
    float*  mx     = (float*)alloc((size_t)N * 8 * 4);
    float*  sbuf   = (float*)alloc((size_t)N * 8 * 4);
    __bf16* Wqkv_b = (__bf16*)alloc((size_t)512 * 1536 * 2);
    __bf16* Wo_b   = (__bf16*)alloc((size_t)512 * 512 * 2);
    __bf16* W1_b   = (__bf16*)alloc((size_t)512 * 2048 * 2);
    __bf16* W2_b   = (__bf16*)alloc((size_t)2048 * 512 * 2);

    auto cvt = [&](const float* in, __bf16* out, int n) {
        f32_to_bf16_kernel<<<(n / 4 + 255) / 256, 256, 0, stream>>>(in, out, n);
    };
    cvt(Wqkv, Wqkv_b, 512 * 1536);
    cvt(Wo,   Wo_b,   512 * 512);
    cvt(W1,   W1_b,   512 * 2048);
    cvt(W2,   W2_b,   2048 * 512);

    // LN1
    ln_kernel<<<(N + 7) / 8, 256, 0, stream>>>(node, ln1g, ln1b, h_b, N);

    // qkv = h @ Wqkv + bqkv  (bf16 out)
    gemm_wmma_kernel<false, false, true>
        <<<dim3(1536 / 128, (N + 63) / 64), 256, 0, stream>>>(
            h_b, Wqkv_b, bqkv, nullptr, qkv_b, N, 512, 1536);

    // segment-softmax state init
    init_kernel<<<(N * 512 + 255) / 256, 256, 0, stream>>>(mx, sbuf, agg,
                                                           N * 8, N * 512);
    // edge logits + segment max
    edge_logits_kernel<<<(E + 7) / 8, 256, 0, stream>>>(qkv_b, src, dst, dattn,
                                                        pattn, abuf, mx, E);
    // exp + segment sum
    edge_exp_kernel<<<(E * 8 + 255) / 256, 256, 0, stream>>>(abuf, dst, mx,
                                                             sbuf, E * 8);
    // weighted scatter of v
    edge_agg_kernel<<<(E + 7) / 8, 256, 0, stream>>>(qkv_b, src, dst, abuf,
                                                     sbuf, agg, E);

    // h1 = agg @ Wo + bo + node   (fp32 A, fp32 out)
    gemm_wmma_kernel<true, false, false>
        <<<dim3(512 / 128, (N + 63) / 64), 256, 0, stream>>>(
            agg, Wo_b, bo, node, h1, N, 512, 512);

    // LN2
    ln_kernel<<<(N + 7) / 8, 256, 0, stream>>>(h1, ln2g, ln2b, h_b, N);

    // t = gelu(h2 @ W1 + b1)  (bf16 out)
    gemm_wmma_kernel<false, true, true>
        <<<dim3(2048 / 128, (N + 63) / 64), 256, 0, stream>>>(
            h_b, W1_b, b1, nullptr, t_b, N, 512, 2048);

    // out = t @ W2 + b2 + h1  (fp32 out -> d_out)
    gemm_wmma_kernel<false, false, false>
        <<<dim3(512 / 128, (N + 63) / 64), 256, 0, stream>>>(
            t_b, W2_b, b2, h1, (float*)d_out, N, 2048, 512);
}
